// SAGPoolHModel_44598940402219
// MI455X (gfx1250) — matrix operand order repaired and checked by reading, not compiled, not run on hardware
//
#include <hip/hip_runtime.h>
#include <hip/hip_bf16.h>
#include <math.h>

#define B_G   128
#define E_TOT (128 * 512 * 16)
#define HIDW  128

typedef __attribute__((ext_vector_type(16))) _Float16 v16h;
typedef __attribute__((ext_vector_type(8)))  _Float16 v8h;
typedef __attribute__((ext_vector_type(8)))  float    v8f;

// ---------------- elementwise / init ----------------
__global__ void k_fill(float* __restrict__ p, float v, int n) {
  int i = blockIdx.x * blockDim.x + threadIdx.x;
  if (i < n) p[i] = v;
}

__global__ void k_cvt_f16(const float* __restrict__ s, _Float16* __restrict__ d, int n) {
  int i = blockIdx.x * blockDim.x + threadIdx.x;
  if (i < n) d[i] = (_Float16)s[i];
}

// W [K,128] row-major f32  ->  Wt [128,K] col-major f16
__global__ void k_wt_cvt(const float* __restrict__ W, _Float16* __restrict__ Wt, int K) {
  int t = blockIdx.x * blockDim.x + threadIdx.x;
  if (t >= K * HIDW) return;
  int k = t >> 7, c = t & 127;
  Wt[(size_t)c * K + k] = (_Float16)W[t];
}

__global__ void k_deg_scatter(const int* __restrict__ dst, const float* __restrict__ w,
                              float* __restrict__ deg, int E) {
  int e = blockIdx.x * blockDim.x + threadIdx.x;
  if (e >= E) return;
  float we = w[e];
  if (we != 0.f) atomicAdd(&deg[dst[e]], we);
}

__global__ void k_dinv(const float* __restrict__ deg, float* __restrict__ dinv, int n) {
  int i = blockIdx.x * blockDim.x + threadIdx.x;
  if (i < n) dinv[i] = rsqrtf(fmaxf(deg[i], 1e-12f));
}

// ---------------- WMMA dense transform: HW = Hh @ Wt^T ------------------------------------
// Hh: [N,KDIM] row-major f16.  Wt: [128,KDIM] col-major f16 (i.e. W transposed).
template <int KDIM>
__global__ __launch_bounds__(256) void k_gcn_matmul(const _Float16* __restrict__ Hh,
                                                    const _Float16* __restrict__ Wt,
                                                    float* __restrict__ HW) {
  const int wave = threadIdx.x >> 5;   // 8 waves: each owns a 16-col tile
  const int lane = threadIdx.x & 31;
  const int half = lane >> 4;
  const int rc   = lane & 15;
  const int row  = blockIdx.x * 16 + rc;   // A row for this lane
  const int col  = wave * 16 + rc;         // B/D col for this lane
  v8f acc = {};
  const _Float16* arow = Hh + (size_t)row * KDIM;
  const _Float16* bcol = Wt + (size_t)col * KDIM;
#pragma unroll
  for (int k0 = 0; k0 < KDIM; k0 += 32) {
    // A fragment: elems 0..7 <- K = k0+8*half+i ; elems 8..15 <- K = k0+16+8*half+i
    v8h a0 = *(const v8h*)(arow + k0 + 8 * half);
    v8h a1 = *(const v8h*)(arow + k0 + 16 + 8 * half);
    // B fragment: elems 0..7 <- K = k0+8*half+i ; elems 8..15 <- K = k0+16+8*half+i
    v8h b0 = *(const v8h*)(bcol + k0 + 8 * half);
    v8h b1 = *(const v8h*)(bcol + k0 + 16 + 8 * half);
    v16h a, b;
#pragma unroll
    for (int i = 0; i < 8; ++i) {
      a[i] = a0[i]; a[i + 8] = a1[i];
      b[i] = b0[i]; b[i + 8] = b1[i];
    }
    acc = __builtin_amdgcn_wmma_f32_16x16x32_f16(false, a, false, b, (short)0, acc,
                                                 false, false);
  }
#pragma unroll
  for (int r = 0; r < 8; ++r)
    HW[(size_t)(blockIdx.x * 16 + r + 8 * half) * HIDW + col] = acc[r];
}

// ---------------- GCN aggregation: out = D^-1/2 A D^-1/2 (hw) + b  ------------------------
__global__ void k_out_init(const float* __restrict__ hw, const float* __restrict__ dinv,
                           const float* __restrict__ bias, float* __restrict__ out, int N) {
  int t = blockIdx.x * blockDim.x + threadIdx.x;
  if (t >= N * HIDW) return;
  int i = t >> 7, f = t & 127;
  float di = dinv[i];
  out[t] = di * di * hw[t] + bias[f];   // self-loop term + bias
}

__global__ void k_edge_scatter128(const int* __restrict__ src, const int* __restrict__ dst,
                                  const float* __restrict__ w, const float* __restrict__ dinv,
                                  const float* __restrict__ hw, float* __restrict__ out, int E) {
  int t = blockIdx.x * blockDim.x + threadIdx.x;
  int e = t >> 5, lane = t & 31;
  if (e >= E) return;
  float we = w[e];
  if (we == 0.f) return;                  // dead edges contribute exactly nothing
  int s = src[e], d = dst[e];
  float nrm = dinv[s] * we * dinv[d];
  const float4 hv = ((const float4*)(hw + (size_t)s * HIDW))[lane];
  float* o = out + (size_t)d * HIDW + lane * 4;
  atomicAdd(o + 0, nrm * hv.x);
  atomicAdd(o + 1, nrm * hv.y);
  atomicAdd(o + 2, nrm * hv.z);
  atomicAdd(o + 3, nrm * hv.w);
}

// ---------------- score GCN (HID -> 1); relu applied at read ------------------------------
__global__ void k_score_matvec(const float* __restrict__ hpre, const float* __restrict__ Ws,
                               float* __restrict__ hws, int N) {
  int i = blockIdx.x * blockDim.x + threadIdx.x;
  if (i >= N) return;
  const float4* hp = (const float4*)(hpre + (size_t)i * HIDW);
  const float4* wp = (const float4*)Ws;
  float acc = 0.f;
#pragma unroll 8
  for (int j = 0; j < 32; ++j) {
    float4 h4 = hp[j], w4 = wp[j];
    acc += fmaxf(h4.x, 0.f) * w4.x + fmaxf(h4.y, 0.f) * w4.y +
           fmaxf(h4.z, 0.f) * w4.z + fmaxf(h4.w, 0.f) * w4.w;
  }
  hws[i] = acc;
}

__global__ void k_score_init(const float* __restrict__ hws, const float* __restrict__ dinv,
                             const float* __restrict__ bs, float* __restrict__ sc, int N) {
  int i = blockIdx.x * blockDim.x + threadIdx.x;
  if (i < N) { float di = dinv[i]; sc[i] = di * di * hws[i] + bs[0]; }
}

__global__ void k_score_scatter(const int* __restrict__ src, const int* __restrict__ dst,
                                const float* __restrict__ w, const float* __restrict__ dinv,
                                const float* __restrict__ hws, float* __restrict__ sc, int E) {
  int e = blockIdx.x * blockDim.x + threadIdx.x;
  if (e >= E) return;
  float we = w[e];
  if (we == 0.f) return;
  int s = src[e], d = dst[e];
  atomicAdd(&sc[d], dinv[s] * we * dinv[d] * hws[s]);
}

// ---------------- per-graph top-k (bitonic, desc score / asc index) -----------------------
__device__ __forceinline__ bool better(float a, int ia, float b, int ib) {
  return (a > b) || (a == b && ia < ib);
}

__global__ __launch_bounds__(256) void k_topk(const float* __restrict__ sc, int n, int k,
                                              float* __restrict__ vals, int* __restrict__ permold,
                                              int* __restrict__ new_id, int* __restrict__ keep) {
  __shared__ float ss[512];
  __shared__ int   si[512];
  int g = blockIdx.x, tid = threadIdx.x;
  for (int i = tid; i < n; i += 256) { ss[i] = tanhf(sc[g * n + i]); si[i] = i; }
  for (int i = tid; i < n; i += 256) { new_id[g * n + i] = 0; keep[g * n + i] = 0; }
  __syncthreads();
  for (int size = 2; size <= n; size <<= 1) {
    for (int stride = size >> 1; stride > 0; stride >>= 1) {
      for (int i = tid; i < n; i += 256) {
        int j = i ^ stride;
        if (j > i) {
          bool up = ((i & size) == 0);
          float a = ss[i], b = ss[j];
          int ia = si[i], ib = si[j];
          bool sw = up ? better(b, ib, a, ia) : better(a, ia, b, ib);
          if (sw) { ss[i] = b; ss[j] = a; si[i] = ib; si[j] = ia; }
        }
      }
      __syncthreads();
    }
  }
  for (int r = tid; r < k; r += 256) {
    float v  = ss[r];
    int  old = g * n + si[r];
    int  j   = g * k + r;              // new global node id = rank order
    vals[j] = v;
    permold[j] = old;
    new_id[old] = j;
    keep[old] = 1;
  }
}

// ---------------- gather + gate (relu at read); also emits f16 copy for next WMMA ---------
__global__ void k_gather(const float* __restrict__ hpre, const int* __restrict__ permold,
                         const float* __restrict__ vals, float* __restrict__ h2,
                         _Float16* __restrict__ h2h, int Nk) {
  int t = blockIdx.x * blockDim.x + threadIdx.x;
  if (t >= Nk * HIDW) return;
  int j = t >> 7, f = t & 127;
  float v = fmaxf(hpre[(size_t)permold[j] * HIDW + f], 0.f) * vals[j];
  h2[t]  = v;
  h2h[t] = (_Float16)v;
}

// ---------------- per-graph (mean || max) readout, accumulated over levels ----------------
__global__ __launch_bounds__(128) void k_readout(const float* __restrict__ h2, int k,
                                                 float* __restrict__ ro) {
  int g = blockIdx.x, f = threadIdx.x;
  float s = 0.f, m = -INFINITY;
  for (int r = 0; r < k; ++r) {
    float v = h2[(size_t)(g * k + r) * HIDW + f];
    s += v;
    m = fmaxf(m, v);
  }
  ro[g * 256 + f]       += s / (float)k;
  ro[g * 256 + 128 + f] += m;
}

// ---------------- edge relabel for next level ---------------------------------------------
__global__ void k_relabel(const int* __restrict__ src, const int* __restrict__ dst,
                          const float* __restrict__ w, const int* __restrict__ keep,
                          const int* __restrict__ new_id, int* __restrict__ src2,
                          int* __restrict__ dst2, float* __restrict__ w2, int E) {
  int e = blockIdx.x * blockDim.x + threadIdx.x;
  if (e >= E) return;
  int s = src[e], d = dst[e];
  bool valid = (keep[s] != 0) && (keep[d] != 0);
  src2[e] = valid ? new_id[s] : 0;
  dst2[e] = valid ? new_id[d] : 0;
  w2[e]   = valid ? w[e] : 0.f;
}

// ---------------- final MLP head -----------------------------------------------------------
__global__ __launch_bounds__(128) void k_mlp(const float* __restrict__ ro,
                                             const float* __restrict__ D1w, const float* __restrict__ D1b,
                                             const float* __restrict__ D2w, const float* __restrict__ D2b,
                                             const float* __restrict__ D3w, const float* __restrict__ D3b,
                                             float* __restrict__ out) {
  __shared__ float s0[256], s1[128], s2[64];
  int g = blockIdx.x, tid = threadIdx.x;
  s0[tid]       = ro[g * 256 + tid];
  s0[tid + 128] = ro[g * 256 + 128 + tid];
  __syncthreads();
  float a1 = D1b[tid];
  for (int i = 0; i < 256; ++i) a1 += s0[i] * D1w[i * 128 + tid];
  s1[tid] = fmaxf(a1, 0.f);
  __syncthreads();
  if (tid < 64) {
    float a2 = D2b[tid];
    for (int i = 0; i < 128; ++i) a2 += s1[i] * D2w[i * 64 + tid];
    s2[tid] = fmaxf(a2, 0.f);
  }
  __syncthreads();
  if (tid < 8) {
    float a3 = D3b[tid];
    for (int i = 0; i < 64; ++i) a3 += s2[i] * D3w[i * 8 + tid];
    out[g * 8 + tid] = a3;
  }
}

// ==========================================================================================
static inline int cdiv(long long a, int b) { return (int)((a + b - 1) / b); }

extern "C" void kernel_launch(void* const* d_in, const int* in_sizes, int n_in,
                              void* d_out, int out_size, void* d_ws, size_t ws_size,
                              hipStream_t stream) {
  (void)in_sizes; (void)n_in; (void)out_size; (void)ws_size;
  const float* x   = (const float*)d_in[0];
  const float* ew  = (const float*)d_in[1];
  const float* Wg[3] = { (const float*)d_in[2],  (const float*)d_in[6],  (const float*)d_in[10] };
  const float* bg[3] = { (const float*)d_in[3],  (const float*)d_in[7],  (const float*)d_in[11] };
  const float* Ws[3] = { (const float*)d_in[4],  (const float*)d_in[8],  (const float*)d_in[12] };
  const float* bs[3] = { (const float*)d_in[5],  (const float*)d_in[9],  (const float*)d_in[13] };
  const float* D1w = (const float*)d_in[14]; const float* D1b = (const float*)d_in[15];
  const float* D2w = (const float*)d_in[16]; const float* D2b = (const float*)d_in[17];
  const float* D3w = (const float*)d_in[18]; const float* D3b = (const float*)d_in[19];
  const int* src0 = (const int*)d_in[20];
  const int* dst0 = (const int*)d_in[21];

  char* ws = (char*)d_ws;
  const size_t MB = 1ull << 20;
  float*    hw    = (float*)   (ws + 0 * MB);    // 32 MB  [N,128] dense-transform output
  float*    outb  = (float*)   (ws + 32 * MB);   // 32 MB  [N,128] aggregated GCN out (pre-relu)
  float*    h2    = (float*)   (ws + 64 * MB);   // 16 MB  gathered+gated features (f32)
  int*      srcA  = (int*)     (ws + 80 * MB);   // ping-pong edge set 0
  int*      dstA  = (int*)     (ws + 84 * MB);
  float*    wA    = (float*)   (ws + 88 * MB);
  int*      srcB  = (int*)     (ws + 92 * MB);   // ping-pong edge set 1
  int*      dstB  = (int*)     (ws + 96 * MB);
  float*    wB    = (float*)   (ws + 100 * MB);
  char*     sm    = ws + 104 * MB;               // small per-node arrays
  float*    deg     = (float*)(sm + 0x000000);
  float*    dinv    = (float*)(sm + 0x040000);
  float*    hws     = (float*)(sm + 0x080000);
  float*    sc      = (float*)(sm + 0x0C0000);
  float*    vals    = (float*)(sm + 0x100000);
  int*      permold = (int*)  (sm + 0x120000);
  int*      new_id  = (int*)  (sm + 0x140000);
  int*      keep    = (int*)  (sm + 0x180000);
  float*    ro      = (float*)(sm + 0x1C0000);
  _Float16* hf16    = (_Float16*)(ws + 108 * MB); // 16 MB  f16 copy of current h (WMMA A)
  _Float16* wt      = (_Float16*)(ws + 124 * MB); // 32 KB  f16 col-major W (WMMA B)

  const int T = 256;
  k_fill<<<cdiv(B_G * 256, T), T, 0, stream>>>(ro, 0.f, B_G * 256);
  k_cvt_f16<<<cdiv(B_G * 512 * 64, T), T, 0, stream>>>(x, hf16, B_G * 512 * 64);

  const int* src = src0;
  const int* dst = dst0;
  const float* w = ew;
  int n = 512;

  for (int L = 0; L < 3; ++L) {
    const int N   = B_G * n;
    const int k   = n / 2;
    const int Nk  = B_G * k;
    const int Kin = (L == 0) ? 64 : 128;

    // degree (self-loop weight 1) + dinv
    k_fill<<<cdiv(N, T), T, 0, stream>>>(deg, 1.f, N);
    k_deg_scatter<<<cdiv(E_TOT, T), T, 0, stream>>>(dst, w, deg, E_TOT);
    k_dinv<<<cdiv(N, T), T, 0, stream>>>(deg, dinv, N);

    // hw = h @ Wg   (f16 operands, WMMA f32 accumulate)
    k_wt_cvt<<<cdiv(Kin * HIDW, T), T, 0, stream>>>(Wg[L], wt, Kin);
    if (L == 0) k_gcn_matmul<64><<<N / 16, 256, 0, stream>>>(hf16, wt, hw);
    else        k_gcn_matmul<128><<<N / 16, 256, 0, stream>>>(hf16, wt, hw);

    // out = self-loop + bias, then scatter edge contributions (L2 atomics)
    k_out_init<<<cdiv((long long)N * HIDW, T), T, 0, stream>>>(hw, dinv, bg[L], outb, N);
    k_edge_scatter128<<<cdiv((long long)E_TOT * 32, T), T, 0, stream>>>(src, dst, w, dinv, hw,
                                                                        outb, E_TOT);

    // score GCN (relu applied at read), then tanh+topk
    k_score_matvec<<<cdiv(N, T), T, 0, stream>>>(outb, Ws[L], hws, N);
    k_score_init<<<cdiv(N, T), T, 0, stream>>>(hws, dinv, bs[L], sc, N);
    k_score_scatter<<<cdiv(E_TOT, T), T, 0, stream>>>(src, dst, w, dinv, hws, sc, E_TOT);
    k_topk<<<B_G, 256, 0, stream>>>(sc, n, k, vals, permold, new_id, keep);

    // gather + gate (also produces next layer's f16 A operand); readout accumulate
    k_gather<<<cdiv((long long)Nk * HIDW, T), T, 0, stream>>>(outb, permold, vals, h2, hf16, Nk);
    k_readout<<<B_G, 128, 0, stream>>>(h2, k, ro);

    // relabel edges for next level (skip after last level)
    if (L == 0) {
      k_relabel<<<cdiv(E_TOT, T), T, 0, stream>>>(src, dst, w, keep, new_id, srcA, dstA, wA, E_TOT);
      src = srcA; dst = dstA; w = wA;
    } else if (L == 1) {
      k_relabel<<<cdiv(E_TOT, T), T, 0, stream>>>(src, dst, w, keep, new_id, srcB, dstB, wB, E_TOT);
      src = srcB; dst = dstB; w = wB;
    }
    n = k;
  }

  k_mlp<<<B_G, 128, 0, stream>>>(ro, D1w, D1b, D2w, D2b, D3w, D3b, (float*)d_out);
}
